// FrequencySurrogate_19868518711893
// MI455X (gfx1250) — matrix operand order repaired and checked by reading, not compiled, not run on hardware
//
#include <hip/hip_runtime.h>
#include <hip/hip_bf16.h>

typedef __attribute__((ext_vector_type(16))) _Float16 v16h;
typedef __attribute__((ext_vector_type(8)))  _Float16 v8h;
typedef __attribute__((ext_vector_type(8)))  float    v8f;

#define NG     256
#define NPER   48
#define NN     (NG*NPER)     // 12288
#define H      128
#define DD     48
#define AF     64
#define SD     512
#define PPG    1128          // C(48,2)
#define CHUNKS 9             // ceil(1128/128)
#define K0P    448           // 3H+D=432 padded to mult of 32
#define FEATLD 456           // LDS feat row stride (halves), 912B, 16B aligned
#define ACTLD  136           // f16 act stride, 272B, 16B aligned
#define BSLD   40            // B-stage row stride (halves), 80B: conflict-free
#define LN_EPS 1e-5f

// ---- WMMA fragment loaders (ISA 16-bit layouts, wave32) -------------------

// A 16x32 f16: lane l -> row m=l&15; halves j<8: k=hi*8+j; j>=8: k=16+hi*8+(j-8)
__device__ __forceinline__ v16h load_a_lds(const _Float16* base, int ld,
                                           int row0, int k0, int lane) {
  const int m = lane & 15, hi = lane >> 4;
  const _Float16* p = base + (row0 + m) * ld + k0 + hi * 8;
  v16h a;
  ((uint4*)&a)[0] = *(const uint4*)(p);       // halves 0..7
  ((uint4*)&a)[1] = *(const uint4*)(p + 16);  // halves 8..15
  return a;
}

// B 32x16 f16 from W^T [Nout, ldk] in GLOBAL memory (small kernels)
__device__ __forceinline__ v16h load_b_wt(const _Float16* wt, int ldk,
                                          int ct, int k0, int lane) {
  const _Float16* p = wt + (ct * 16 + (lane & 15)) * ldk + k0 + (lane >> 4) * 8;
  v16h b;
  ((uint4*)&b)[0] = *(const uint4*)(p);
  ((uint4*)&b)[1] = *(const uint4*)(p + 16);
  return b;
}

// B 32x16 f16 from LDS stage buffer [128][BSLD] (k-major within row)
__device__ __forceinline__ v16h load_b_lds(const _Float16* bst, int ct, int lane) {
  const _Float16* p = bst + (ct * 16 + (lane & 15)) * BSLD + (lane >> 4) * 8;
  v16h b;
  ((uint4*)&b)[0] = *(const uint4*)(p);
  ((uint4*)&b)[1] = *(const uint4*)(p + 16);
  return b;
}

__device__ __forceinline__ v8f wmma_f16(v16h a, v16h b, v8f c) {
  return __builtin_amdgcn_wmma_f32_16x16x32_f16(false, a, false, b,
                                                (short)0, c, false, false);
}

__device__ __forceinline__ v8h ld8h(const _Float16* p) {
  v8h x; *(uint4*)&x = *(const uint4*)p; return x;
}
__device__ __forceinline__ void st8h(_Float16* p, v8h x) {
  *(uint4*)p = *(uint4*)&x;
}

// ---- async global -> LDS staging of one 32(k) x 128(n) B tile -------------
// 256 threads: thread t covers row n=t>>1, 16-half slice part=t&1.
// INST_OFFSET applies to both LDS and global addresses (ISA async LDS ops).
__device__ __forceinline__ void stage_b(const _Float16* WT, int ldW, int ks,
                                        _Float16* dst, int tid) {
  const int n = tid >> 1, part = tid & 1;
  const uint32_t voff = (uint32_t)((n * ldW + ks * 32 + part * 16) * 2);
  const uint32_t lds  = (uint32_t)(uintptr_t)(dst + n * BSLD + part * 16);
  asm volatile("global_load_async_to_lds_b128 %0, %1, %2 offset:0\n\t"
               "global_load_async_to_lds_b128 %0, %1, %2 offset:16"
               :
               : "v"(lds), "v"(voff), "s"(WT)
               : "memory");
}

__device__ __forceinline__ void wait_async0() {
  asm volatile("s_wait_asynccnt 0x0" ::: "memory");
}

// GEMM over a 16-row tile with double-buffered LDS B staging.
template <int NSTEPS>
__device__ __forceinline__ void gemm_staged(const _Float16* A, int ldA, int row0,
                                            const _Float16* WT, int ldW,
                                            _Float16* bst0, _Float16* bst1,
                                            v8f acc[8], int tid, int lane) {
  stage_b(WT, ldW, 0, bst0, tid);
#pragma unroll
  for (int ks = 0; ks < NSTEPS; ++ks) {
    _Float16* cur = (ks & 1) ? bst1 : bst0;
    _Float16* nxt = (ks & 1) ? bst0 : bst1;
    wait_async0();        // own async stores to LDS done
    __syncthreads();      // all waves' stores published; prev buffer consumers done
    if (ks + 1 < NSTEPS) stage_b(WT, ldW, ks + 1, nxt, tid);
    v16h a = load_a_lds(A, ldA, row0, ks * 32, lane);
#pragma unroll
    for (int ct = 0; ct < 8; ++ct)
      acc[ct] = wmma_f16(a, load_b_lds(cur, ct, lane), acc[ct]);
  }
}

// Register-space bias + LayerNorm + SiLU on the C-fragment layout:
// lane holds cols n=ct*16+nlo of rows i+8*hi; a row's 128 values live in one
// 16-lane half -> butterfly shfl_xor(1,2,4,8) yields per-row mean/var.
__device__ __forceinline__ void bias_ln_silu_reg(v8f acc[8], const float* bias,
                                                 const float* gamma,
                                                 const float* beta,
                                                 _Float16* act, int wave, int lane) {
  const int nlo = lane & 15, hi = lane >> 4;
#pragma unroll
  for (int ct = 0; ct < 8; ++ct) {
    const float b = bias[ct * 16 + nlo];
#pragma unroll
    for (int i = 0; i < 8; ++i) acc[ct][i] += b;
  }
  float s[8], ss[8];
#pragma unroll
  for (int i = 0; i < 8; ++i) { s[i] = 0.f; ss[i] = 0.f; }
#pragma unroll
  for (int ct = 0; ct < 8; ++ct)
#pragma unroll
    for (int i = 0; i < 8; ++i) {
      float x = acc[ct][i]; s[i] += x; ss[i] += x * x;
    }
#pragma unroll
  for (int m = 1; m < 16; m <<= 1) {
#pragma unroll
    for (int i = 0; i < 8; ++i) {
      s[i]  += __shfl_xor(s[i],  m, 32);
      ss[i] += __shfl_xor(ss[i], m, 32);
    }
  }
  float mean[8], inv[8];
#pragma unroll
  for (int i = 0; i < 8; ++i) {
    mean[i] = s[i] * (1.f / 128.f);
    float var = ss[i] * (1.f / 128.f) - mean[i] * mean[i];
    inv[i] = rsqrtf(var + LN_EPS);
  }
#pragma unroll
  for (int ct = 0; ct < 8; ++ct) {
    const int n = ct * 16 + nlo;
    const float gm = gamma[n], bt = beta[n];
#pragma unroll
    for (int i = 0; i < 8; ++i) {
      float y = (acc[ct][i] - mean[i]) * inv[i] * gm + bt;
      y = y * (1.f / (1.f + __expf(-y)));   // SiLU
      act[(wave * 16 + i + 8 * hi) * ACTLD + n] = (_Float16)y;
    }
  }
}

// ---- prep: W f32 [K,N] -> W^T f16 [N,KP] (zero padded) --------------------
__global__ void __launch_bounds__(256) wtrans_kernel(const float* __restrict__ W,
                                                     _Float16* __restrict__ WT,
                                                     int K, int Nn, int KP) {
  int idx = blockIdx.x * 256 + threadIdx.x;
  if (idx >= Nn * KP) return;
  int n = idx / KP, k = idx % KP;
  WT[idx] = (k < K) ? (_Float16)W[(long)k * Nn + n] : (_Float16)0.f;
}

// ---- node embedding: [N,64]x[64,128]+b -> f16 -----------------------------
__global__ void __launch_bounds__(256) node_emb_kernel(
    const float* __restrict__ node_h, const _Float16* __restrict__ atom_wT,
    const float* __restrict__ atom_b, _Float16* __restrict__ node_emb) {
  const int wave = threadIdx.x >> 5, lane = threadIdx.x & 31;
  const int m = lane & 15, hi = lane >> 4, nlo = lane & 15;
  const int row0 = blockIdx.x * 16;
  const float* rowp = node_h + (long)(row0 + m) * AF;
  v8f acc = {};
#pragma unroll
  for (int ks = 0; ks < 2; ++ks) {
    const int k0 = ks * 32;
    v16h a;
#pragma unroll
    for (int j = 0; j < 8; ++j) a[j]     = (_Float16)rowp[k0 + hi * 8 + j];
#pragma unroll
    for (int j = 0; j < 8; ++j) a[8 + j] = (_Float16)rowp[k0 + 16 + hi * 8 + j];
    acc = wmma_f16(a, load_b_wt(atom_wT, AF, wave, k0, lane), acc);
  }
  const float b = atom_b[wave * 16 + nlo];
#pragma unroll
  for (int i = 0; i < 8; ++i)
    node_emb[(long)(row0 + i + 8 * hi) * H + wave * 16 + nlo] =
        (_Float16)(acc[i] + b);
}

// ---- fused pair MLP + segment sum -----------------------------------------
__global__ void __launch_bounds__(256) pair_kernel(
    const _Float16* __restrict__ node_emb, const float* __restrict__ pos,
    const int* __restrict__ left, const int* __restrict__ right,
    const _Float16* __restrict__ pw0T, const float* __restrict__ pb0,
    const float* __restrict__ pg0, const float* __restrict__ pbe0,
    const _Float16* __restrict__ pw1T, const float* __restrict__ pb1,
    const float* __restrict__ pg1, const float* __restrict__ pbe1,
    const _Float16* __restrict__ pw2T, const float* __restrict__ pb2,
    float* __restrict__ graph_h) {
  __shared__ __align__(16) union {
    _Float16 feat[128 * FEATLD];   // 116736 B (L0 input)
    _Float16 act[128 * ACTLD];     //  34816 B (L1/L2 input, overlays feat)
  } u;
  __shared__ __align__(16) _Float16 bst[2][128 * BSLD];   // 20480 B
  __shared__ float gsum[H];

  const int tid = threadIdx.x;
  const int wave = tid >> 5, lane = tid & 31;
  const int g = blockIdx.x / CHUNKS, ch = blockIdx.x % CHUNKS;

  if (tid < H) gsum[tid] = 0.f;

  // --- build pair features [128, 448] f16 in LDS (vectorized) ---
  {
    const int r = tid >> 1, hh = tid & 1;
    const int pr = ch * 128 + r;
    _Float16* frow = u.feat + r * FEATLD;
    if (pr < PPG) {
      const int pid = g * PPG + pr;
      const int li = left[pid], ri = right[pid];
      const _Float16* el = node_emb + li * H;
      const _Float16* er = node_emb + ri * H;
      if (hh == 0) {
#pragma unroll
        for (int j = 0; j < 16; ++j) st8h(frow + j * 8, ld8h(el + j * 8));
#pragma unroll
        for (int j = 0; j < 12; ++j) st8h(frow + 128 + j * 8, ld8h(er + j * 8));
      } else {
#pragma unroll
        for (int j = 12; j < 16; ++j) st8h(frow + 128 + j * 8, ld8h(er + j * 8));
#pragma unroll
        for (int j = 0; j < 16; ++j) {
          v8h a = ld8h(el + j * 8), b = ld8h(er + j * 8);
          st8h(frow + 256 + j * 8, a * b);            // v_pk_mul_f16
        }
        float dx = pos[li * 3 + 0] - pos[ri * 3 + 0];
        float dy = pos[li * 3 + 1] - pos[ri * 3 + 1];
        float dz = pos[li * 3 + 2] - pos[ri * 3 + 2];
        float dist = sqrtf(dx * dx + dy * dy + dz * dz);
        const float spacing = 12.0f / 47.0f;
        const float coeff = -0.5f / (spacing * spacing);
#pragma unroll
        for (int j = 0; j < DD; ++j) {
          float d = dist - spacing * (float)j;
          frow[384 + j] = (_Float16)__expf(coeff * d * d);
        }
        v8h z = {};
        st8h(frow + 432, z); st8h(frow + 440, z);
      }
    } else {
      v8h z = {};
      if (hh == 0) {
#pragma unroll
        for (int j = 0; j < 28; ++j) st8h(frow + j * 8, z);
      } else {
#pragma unroll
        for (int j = 28; j < 56; ++j) st8h(frow + j * 8, z);
      }
    }
  }
  __syncthreads();

  v8f zero = {};
  v8f acc[8];

  // --- layer 0: [16,448] x [448,128] ---
#pragma unroll
  for (int ct = 0; ct < 8; ++ct) acc[ct] = zero;
  gemm_staged<14>(u.feat, FEATLD, wave * 16, pw0T, K0P, bst[0], bst[1],
                  acc, tid, lane);
  __syncthreads();    // all feat reads done -> act may overlay feat
  bias_ln_silu_reg(acc, pb0, pg0, pbe0, u.act, wave, lane);

  // --- layer 1 (act rows are wave-private; no extra barriers needed) ---
#pragma unroll
  for (int ct = 0; ct < 8; ++ct) acc[ct] = zero;
  gemm_staged<4>(u.act, ACTLD, wave * 16, pw1T, H, bst[0], bst[1],
                 acc, tid, lane);
  bias_ln_silu_reg(acc, pb1, pg1, pbe1, u.act, wave, lane);

  // --- layer 2 + masked segment-sum ---
#pragma unroll
  for (int ct = 0; ct < 8; ++ct) acc[ct] = zero;
  gemm_staged<4>(u.act, ACTLD, wave * 16, pw2T, H, bst[0], bst[1],
                 acc, tid, lane);
  {
    const int nlo = lane & 15, hi = lane >> 4;
#pragma unroll
    for (int ct = 0; ct < 8; ++ct) {
      const int n = ct * 16 + nlo;
      const float b = pb2[n];
      float partial = 0.f;
#pragma unroll
      for (int i = 0; i < 8; ++i) {
        const int prow = ch * 128 + wave * 16 + i + 8 * hi;
        if (prow < PPG) partial += acc[ct][i] + b;
      }
      atomicAdd(&gsum[n], partial);
    }
  }
  __syncthreads();
  if (tid < H) atomicAdd(&graph_h[g * H + tid], gsum[tid]);
}

// ---- out MLP: 128->128->128->512, softplus, row-normalize -----------------
__global__ void __launch_bounds__(256) out_kernel(
    const float* __restrict__ graph_h,
    const _Float16* __restrict__ ow0T, const float* __restrict__ ob0,
    const _Float16* __restrict__ ow1T, const float* __restrict__ ob1,
    const _Float16* __restrict__ ow2T, const float* __restrict__ ob2,
    float* __restrict__ out) {
  __shared__ __align__(16) _Float16 actA[16 * ACTLD];
  __shared__ __align__(16) _Float16 actB[16 * ACTLD];
  __shared__ float spec[16 * 520];
  __shared__ float ssq[16];
  __shared__ float inv_[16];
  const int tid = threadIdx.x, wave = tid >> 5, lane = tid & 31;
  const int nlo = lane & 15, hi = lane >> 4;
  const int g0 = blockIdx.x * 16;

  for (int idx = tid; idx < 16 * H; idx += 256) {
    int rr = idx >> 7, c = idx & 127;
    actA[rr * ACTLD + c] = (_Float16)graph_h[(g0 + rr) * H + c];
  }
  if (tid < 16) ssq[tid] = 0.f;
  __syncthreads();

  {  // layer 0
    v8f acc = {};
#pragma unroll
    for (int ks = 0; ks < 4; ++ks)
      acc = wmma_f16(load_a_lds(actA, ACTLD, 0, ks * 32, lane),
                     load_b_wt(ow0T, H, wave, ks * 32, lane), acc);
    const float b = ob0[wave * 16 + nlo];
#pragma unroll
    for (int i = 0; i < 8; ++i) {
      float y = acc[i] + b;
      y = y * (1.f / (1.f + __expf(-y)));
      actB[(i + 8 * hi) * ACTLD + wave * 16 + nlo] = (_Float16)y;
    }
  }
  __syncthreads();
  {  // layer 1
    v8f acc = {};
#pragma unroll
    for (int ks = 0; ks < 4; ++ks)
      acc = wmma_f16(load_a_lds(actB, ACTLD, 0, ks * 32, lane),
                     load_b_wt(ow1T, H, wave, ks * 32, lane), acc);
    const float b = ob1[wave * 16 + nlo];
#pragma unroll
    for (int i = 0; i < 8; ++i) {
      float y = acc[i] + b;
      y = y * (1.f / (1.f + __expf(-y)));
      actA[(i + 8 * hi) * ACTLD + wave * 16 + nlo] = (_Float16)y;
    }
  }
  __syncthreads();
  // layer 2: 512 cols = 32 tiles, 4 per wave; softplus + sumsq
#pragma unroll
  for (int q = 0; q < 4; ++q) {
    const int ct = wave + q * 8;
    v8f acc = {};
#pragma unroll
    for (int ks = 0; ks < 4; ++ks)
      acc = wmma_f16(load_a_lds(actA, ACTLD, 0, ks * 32, lane),
                     load_b_wt(ow2T, H, ct, ks * 32, lane), acc);
    const int n = ct * 16 + nlo;
    const float b = ob2[n];
#pragma unroll
    for (int i = 0; i < 8; ++i) {
      float y = acc[i] + b;
      float sp = (y > 20.f) ? y : log1pf(__expf(y));
      spec[(i + 8 * hi) * 520 + n] = sp;
      atomicAdd(&ssq[i + 8 * hi], sp * sp);
    }
  }
  __syncthreads();
  if (tid < 16) inv_[tid] = 1.f / fmaxf(sqrtf(ssq[tid]), 1e-12f);
  __syncthreads();
  for (int idx = tid; idx < 16 * SD; idx += 256) {
    int rr = idx >> 9, c = idx & 511;
    out[(g0 + rr) * SD + c] = spec[rr * 520 + c] * inv_[rr];
  }
}

// ---------------------------------------------------------------------------
extern "C" void kernel_launch(void* const* d_in, const int* in_sizes, int n_in,
                              void* d_out, int out_size, void* d_ws, size_t ws_size,
                              hipStream_t stream) {
  (void)in_sizes; (void)n_in; (void)out_size; (void)ws_size;
  const float* node_h = (const float*)d_in[0];
  const float* pos    = (const float*)d_in[1];
  // d_in[2] = batch (unused)
  const int* left  = (const int*)d_in[3];
  const int* right = (const int*)d_in[4];
  // d_in[5] = pair_batch (implicit: pairs are graph-major)
  const float* atom_w = (const float*)d_in[6];
  const float* atom_b = (const float*)d_in[7];
  const float* pw0 = (const float*)d_in[8];
  const float* pb0 = (const float*)d_in[9];
  const float* pg0 = (const float*)d_in[10];
  const float* pbe0 = (const float*)d_in[11];
  const float* pw1 = (const float*)d_in[12];
  const float* pb1 = (const float*)d_in[13];
  const float* pg1 = (const float*)d_in[14];
  const float* pbe1 = (const float*)d_in[15];
  const float* pw2 = (const float*)d_in[16];
  const float* pb2 = (const float*)d_in[17];
  const float* ow0 = (const float*)d_in[18];
  const float* ob0 = (const float*)d_in[19];
  const float* ow1 = (const float*)d_in[20];
  const float* ob1 = (const float*)d_in[21];
  const float* ow2 = (const float*)d_in[22];
  const float* ob2 = (const float*)d_in[23];

  char* w = (char*)d_ws;
  size_t off = 0;
  _Float16* node_emb = (_Float16*)(w + off); off += (size_t)NN * H * 2;
  _Float16* atom_wT  = (_Float16*)(w + off); off += (size_t)H * AF * 2;
  _Float16* pw0T     = (_Float16*)(w + off); off += (size_t)H * K0P * 2;
  _Float16* pw1T     = (_Float16*)(w + off); off += (size_t)H * H * 2;
  _Float16* pw2T     = (_Float16*)(w + off); off += (size_t)H * H * 2;
  _Float16* ow0T     = (_Float16*)(w + off); off += (size_t)H * H * 2;
  _Float16* ow1T     = (_Float16*)(w + off); off += (size_t)H * H * 2;
  _Float16* ow2T     = (_Float16*)(w + off); off += (size_t)SD * H * 2;
  float*    graph_h  = (float*)(w + off);    off += (size_t)NG * H * 4;

  auto T = [&](const float* W, _Float16* WT, int K, int Nn, int KP) {
    int tot = Nn * KP;
    wtrans_kernel<<<(tot + 255) / 256, 256, 0, stream>>>(W, WT, K, Nn, KP);
  };
  T(atom_w, atom_wT, AF, H, AF);
  T(pw0, pw0T, 3 * H + DD, H, K0P);
  T(pw1, pw1T, H, H, H);
  T(pw2, pw2T, H, H, H);
  T(ow0, ow0T, H, H, H);
  T(ow1, ow1T, H, H, H);
  T(ow2, ow2T, H, SD, H);
  hipMemsetAsync(graph_h, 0, (size_t)NG * H * 4, stream);

  node_emb_kernel<<<NN / 16, 256, 0, stream>>>(node_h, atom_wT, atom_b, node_emb);
  pair_kernel<<<NG * CHUNKS, 256, 0, stream>>>(
      node_emb, pos, left, right, pw0T, pb0, pg0, pbe0, pw1T, pb1, pg1, pbe1,
      pw2T, pb2, graph_h);
  out_kernel<<<NG / 16, 256, 0, stream>>>(graph_h, ow0T, ob0, ow1T, ob1, ow2T,
                                          ob2, (float*)d_out);
}
